// AffineExpert_6760278524242
// MI455X (gfx1250) — matrix operand order repaired and checked by reading, not compiled, not run on hardware
//
#include <hip/hip_runtime.h>
#include <hip/hip_bf16.h>

// ---------------------------------------------------------------------------
// AffineExpert fused pipeline for gfx1250 (MI455X, wave32, WMMA)
//   K1: fused GEMM  a = sigmoid(x@Wa^T + ba), u_in = x@Wu^T + bu,
//                   g = x@Wg^T + bg (computed by blockIdx.y==0 blocks,
//                   reusing the x tile already staged in LDS)      (bf16 WMMA)
//   K2: sequential scan over S (16 independent batch scans)
// ---------------------------------------------------------------------------

typedef __attribute__((ext_vector_type(16))) __bf16 v16bf;
typedef __attribute__((ext_vector_type(8)))  __bf16 bf16x8;
typedef __attribute__((ext_vector_type(4)))  __bf16 bf16x4;
typedef __attribute__((ext_vector_type(8)))  float  v8f;
typedef __attribute__((ext_vector_type(4)))  float  f32x4;   // ext-vector for NT loads

#define LDS_STRIDE 40   // 32 bf16 data + 8 pad per row (80B rows, 16B aligned)
#define GOFF (3 * 128 * LDS_STRIDE)            // Wg tile region within a buffer
#define BUF_HALFS ((3 * 128 + 16) * LDS_STRIDE)

static __device__ __forceinline__ v8f vzero8() {
  v8f z;
#pragma unroll
  for (int i = 0; i < 8; ++i) z[i] = 0.0f;
  return z;
}

// Stage a 128x32 f32 tile from global into LDS as bf16 (converted in flight).
static __device__ __forceinline__ void stage_tile(const float* __restrict__ src,
                                                  int row0, int k0, int ld,
                                                  __bf16* __restrict__ dst, int tid) {
#pragma unroll
  for (int i = 0; i < 4; ++i) {
    int idx = (tid << 2) + i;          // 0..1023  (1024 float4 = 128x32 f32)
    int row = idx >> 3;                // 0..127
    int c4  = (idx & 7) << 2;          // 0,4,...,28
    float4 f = *(const float4*)(src + (size_t)(row0 + row) * ld + k0 + c4);
    bf16x4 h;
    h.x = (__bf16)f.x; h.y = (__bf16)f.y; h.z = (__bf16)f.z; h.w = (__bf16)f.w;
    *(bf16x4*)(dst + row * LDS_STRIDE + c4) = h;   // ds_store_b64
  }
}

// Stage the 16x32 Wg tile (only the first 128 threads participate).
static __device__ __forceinline__ void stage_tile_g(const float* __restrict__ src,
                                                    int k0, int ld,
                                                    __bf16* __restrict__ dst, int tid) {
  if (tid < 128) {
    int row = tid >> 3;                // 0..15
    int c4  = (tid & 7) << 2;          // 0,4,...,28
    float4 f = *(const float4*)(src + (size_t)row * ld + k0 + c4);
    bf16x4 h;
    h.x = (__bf16)f.x; h.y = (__bf16)f.y; h.z = (__bf16)f.z; h.w = (__bf16)f.w;
    *(bf16x4*)(dst + row * LDS_STRIDE + c4) = h;
  }
}

// A-operand fragment (16x32 bf16): lanes 0-15 -> K 0..7 | 16..23,
// lanes 16-31 -> K 8..15 | 24..31 (ISA 7.12.2 16-bit A layout).
static __device__ __forceinline__ v16bf fragA_lds(const __bf16* base, int row, int lane) {
  const int kb = (lane >> 4) << 3;                 // 0 or 8
  const __bf16* p = base + row * LDS_STRIDE;
  bf16x8 lo = *(const bf16x8*)(p + kb);            // ds_load_b128
  bf16x8 hi = *(const bf16x8*)(p + 16 + kb);       // ds_load_b128
  v16bf r;
#pragma unroll
  for (int i = 0; i < 8; ++i) { r[i] = lo[i]; r[i + 8] = hi[i]; }
  return r;
}

// B-operand fragment (32x16 bf16, column n = lane&15): lanes 0-15 hold
// K 0..7 | 16..23, lanes 16-31 hold K 8..15 | 24..31 (matches A).
static __device__ __forceinline__ v16bf fragB_lds(const __bf16* base, int row, int lane) {
  const int kb = (lane >> 4) << 3;                 // 0 or 8
  const __bf16* p = base + row * LDS_STRIDE + kb;
  bf16x8 lo = *(const bf16x8*)(p);
  bf16x8 hi = *(const bf16x8*)(p + 16);
  v16bf r;
#pragma unroll
  for (int i = 0; i < 8; ++i) { r[i] = lo[i]; r[i + 8] = hi[i]; }
  return r;
}

// ---------------------------------------------------------------------------
// K1: fused a/u/g GEMM.  M=32768 N=1024 K=1024, 128x128 block tile, 8 waves
// in a 4(M) x 2(N) layout, each wave owns 32x64 (2x4 WMMA tiles) per output.
// Blocks with blockIdx.y==0 additionally compute the g = x@Wg^T slab for
// their 128 rows (wave w -> g rows m0 + w*16), reusing the staged x tile.
// ---------------------------------------------------------------------------
__global__ __launch_bounds__(256) void fused_aug_gemm(
    const float* __restrict__ x,  const float* __restrict__ Wa,
    const float* __restrict__ ba, const float* __restrict__ Wu,
    const float* __restrict__ bu, const float* __restrict__ Wg,
    const float* __restrict__ bg, float* __restrict__ aout,
    float* __restrict__ uout,     float* __restrict__ gout) {
  __shared__ __bf16 lds[2][BUF_HALFS];             // x | Wa | Wu | Wg, dbl-buffered
  const int tid  = threadIdx.x;
  const int lane = tid & 31, wave = tid >> 5;
  const int wm = wave >> 1, wn = wave & 1;
  const int m0 = blockIdx.x * 128;                 // 256 M tiles
  const int n0 = blockIdx.y * 128;                 // 8 N tiles
  const bool do_g = (blockIdx.y == 0);             // block-uniform (EXEC stays full)
  const int D = 1024, N = 1024;

  v8f accA[2][4], accU[2][4], accG;
#pragma unroll
  for (int i = 0; i < 2; ++i)
#pragma unroll
    for (int j = 0; j < 4; ++j) { accA[i][j] = vzero8(); accU[i][j] = vzero8(); }
  accG = vzero8();

  stage_tile(x,  m0, 0, D, lds[0] + 0 * 128 * LDS_STRIDE, tid);
  stage_tile(Wa, n0, 0, D, lds[0] + 1 * 128 * LDS_STRIDE, tid);
  stage_tile(Wu, n0, 0, D, lds[0] + 2 * 128 * LDS_STRIDE, tid);
  if (do_g) stage_tile_g(Wg, 0, D, lds[0] + GOFF, tid);
  __syncthreads();

  for (int kt = 0; kt < 32; ++kt) {
    const int cur = kt & 1;
    if (kt + 1 < 32) {
      const int k0 = (kt + 1) * 32;
      stage_tile(x,  m0, k0, D, lds[cur ^ 1] + 0 * 128 * LDS_STRIDE, tid);
      stage_tile(Wa, n0, k0, D, lds[cur ^ 1] + 1 * 128 * LDS_STRIDE, tid);
      stage_tile(Wu, n0, k0, D, lds[cur ^ 1] + 2 * 128 * LDS_STRIDE, tid);
      if (do_g) stage_tile_g(Wg, k0, D, lds[cur ^ 1] + GOFF, tid);
      if (kt + 2 < 32) {                           // pull K+2 into L2 early
        const int kp = (kt + 2) * 32;
        const size_t pr = (size_t)(tid >> 1) * D + kp + (size_t)(tid & 1) * 16;
        __builtin_prefetch(x  + (size_t)m0 * D + pr, 0, 1);  // global_prefetch_b8
        __builtin_prefetch(Wa + (size_t)n0 * D + pr, 0, 1);
        __builtin_prefetch(Wu + (size_t)n0 * D + pr, 0, 1);
      }
    }
    const __bf16* As  = lds[cur];
    const __bf16* Bas = As + 1 * 128 * LDS_STRIDE;
    const __bf16* Bus = As + 2 * 128 * LDS_STRIDE;
    const __bf16* Gs  = As + GOFF;

    v16bf af[2], fa[4], fu[4];
#pragma unroll
    for (int mi = 0; mi < 2; ++mi)
      af[mi] = fragA_lds(As, wm * 32 + mi * 16 + (lane & 15), lane);
#pragma unroll
    for (int ni = 0; ni < 4; ++ni) {
      fa[ni] = fragB_lds(Bas, wn * 64 + ni * 16 + (lane & 15), lane);
      fu[ni] = fragB_lds(Bus, wn * 64 + ni * 16 + (lane & 15), lane);
    }
#pragma unroll
    for (int mi = 0; mi < 2; ++mi)
#pragma unroll
      for (int ni = 0; ni < 4; ++ni) {
        accA[mi][ni] = __builtin_amdgcn_wmma_f32_16x16x32_bf16(
            false, af[mi], false, fa[ni], (short)0, accA[mi][ni], false, false);
        accU[mi][ni] = __builtin_amdgcn_wmma_f32_16x16x32_bf16(
            false, af[mi], false, fu[ni], (short)0, accU[mi][ni], false, false);
      }
    if (do_g) {
      v16bf ag = fragA_lds(As, wave * 16 + (lane & 15), lane);
      v16bf bgf = fragB_lds(Gs, lane & 15, lane);
      accG = __builtin_amdgcn_wmma_f32_16x16x32_bf16(
          false, ag, false, bgf, (short)0, accG, false, false);
    }
    __syncthreads();
  }

  // Epilogue: C layout = VGPR p -> row p (lanes 0-15) / row 8+p (lanes 16-31),
  // col = lane&15.  Bias + sigmoid for a; NT stores (streamed, no L2 reuse).
#pragma unroll
  for (int mi = 0; mi < 2; ++mi)
#pragma unroll
    for (int ni = 0; ni < 4; ++ni) {
      const int col   = n0 + wn * 64 + ni * 16 + (lane & 15);
      const int rbase = m0 + wm * 32 + mi * 16 + ((lane >> 4) << 3);
      const float bav = ba[col], buv = bu[col];
#pragma unroll
      for (int e = 0; e < 8; ++e) {
        const size_t off = (size_t)(rbase + e) * N + col;
        const float va = accA[mi][ni][e] + bav;
        __builtin_nontemporal_store(1.0f / (1.0f + __expf(-va)), &aout[off]);
        __builtin_nontemporal_store(accU[mi][ni][e] + buv, &uout[off]);
      }
    }
  if (do_g) {
    const int rr = lane & 15;
    const float bgv = bg[rr];
#pragma unroll
    for (int e = 0; e < 8; ++e) {
      const int row = m0 + wave * 16 + e + ((lane >> 4) << 3);
      __builtin_nontemporal_store(accG[e] + bgv, &gout[(size_t)row * 16 + rr]);
    }
  }
}

// ---------------------------------------------------------------------------
// K2: sequential scan.  Block b owns batch b; thread owns 4 h-channels with
// state + its 4 rows of u,v entirely in registers.  Per step: R=16 partial
// dot-products, wave shuffle reduce, 8-wave LDS combine, state update.
// a/u_in are 256MB one-shot streams -> non-temporal 128-bit loads.
// ---------------------------------------------------------------------------
__global__ __launch_bounds__(256) void scan_kernel(
    const float* __restrict__ a, const float* __restrict__ uin,
    const float* __restrict__ g, const float* __restrict__ umat,
    const float* __restrict__ vmat, float* __restrict__ out) {
  const int b = blockIdx.x, tid = threadIdx.x;
  const int lane = tid & 31, wave = tid >> 5;
  const int h0 = tid * 4;
  __shared__ float wsum[8 * 16];
  __shared__ float grs[16];

  float vreg[4][16], ureg[4][16];
#pragma unroll
  for (int i = 0; i < 4; ++i)
#pragma unroll
    for (int q = 0; q < 4; ++q) {
      float4 vv = *(const float4*)(vmat + (size_t)(h0 + i) * 16 + q * 4);
      float4 uu = *(const float4*)(umat + (size_t)(h0 + i) * 16 + q * 4);
      vreg[i][q*4+0]=vv.x; vreg[i][q*4+1]=vv.y; vreg[i][q*4+2]=vv.z; vreg[i][q*4+3]=vv.w;
      ureg[i][q*4+0]=uu.x; ureg[i][q*4+1]=uu.y; ureg[i][q*4+2]=uu.z; ureg[i][q*4+3]=uu.w;
    }

  float st[4] = {0.f, 0.f, 0.f, 0.f};
  const size_t baseBS = (size_t)b * 2048;

  for (int t = 0; t < 2048; ++t) {
    const size_t rowoff = (baseBS + t) * 1024;
    f32x4 a4 = __builtin_nontemporal_load((const f32x4*)(a   + rowoff + h0));
    f32x4 u4 = __builtin_nontemporal_load((const f32x4*)(uin + rowoff + h0));

    float p[16];
#pragma unroll
    for (int r = 0; r < 16; ++r)
      p[r] = st[0]*vreg[0][r] + st[1]*vreg[1][r] + st[2]*vreg[2][r] + st[3]*vreg[3][r];
#pragma unroll
    for (int r = 0; r < 16; ++r) {
      p[r] += __shfl_xor(p[r], 16, 32);
      p[r] += __shfl_xor(p[r], 8, 32);
      p[r] += __shfl_xor(p[r], 4, 32);
      p[r] += __shfl_xor(p[r], 2, 32);
      p[r] += __shfl_xor(p[r], 1, 32);
    }
    if (lane == 0) {
#pragma unroll
      for (int r = 0; r < 16; ++r) wsum[wave * 16 + r] = p[r];
    }
    __syncthreads();
    if (tid < 16) {
      float s = 0.f;
#pragma unroll
      for (int w = 0; w < 8; ++w) s += wsum[w * 16 + tid];
      grs[tid] = s * g[(baseBS + t) * 16 + tid];   // (state@v) * g_t
    }
    __syncthreads();
#pragma unroll
    for (int i = 0; i < 4; ++i) {
      float lr = 0.f;
#pragma unroll
      for (int r = 0; r < 16; ++r) lr += grs[r] * ureg[i][r];
      st[i] = a4[i] * st[i] + u4[i] + lr;          // a_t*state + u_t + lr@u^T
    }
    __syncthreads();
  }
#pragma unroll
  for (int i = 0; i < 4; ++i) out[(size_t)b * 1024 + h0 + i] = st[i];
}

// ---------------------------------------------------------------------------
extern "C" void kernel_launch(void* const* d_in, const int* in_sizes, int n_in,
                              void* d_out, int out_size, void* d_ws, size_t ws_size,
                              hipStream_t stream) {
  const float* x  = (const float*)d_in[0];
  const float* Wa = (const float*)d_in[1];
  const float* ba = (const float*)d_in[2];
  const float* Wg = (const float*)d_in[3];
  const float* bg = (const float*)d_in[4];
  const float* Wu = (const float*)d_in[5];
  const float* bu = (const float*)d_in[6];
  const float* u  = (const float*)d_in[7];
  const float* v  = (const float*)d_in[8];
  float* out = (float*)d_out;

  const size_t MS = (size_t)16 * 2048;             // B*S = 32768
  float* aout = (float*)d_ws;                      // [32768,1024] f32 = 128 MB
  float* uout = aout + MS * 1024;                  // [32768,1024] f32 = 128 MB
  float* gout = uout + MS * 1024;                  // [32768,16]   f32 =   2 MB

  dim3 gemm_grid(256, 8);
  fused_aug_gemm<<<gemm_grid, 256, 0, stream>>>(x, Wa, ba, Wu, bu, Wg, bg,
                                                aout, uout, gout);
  scan_kernel<<<16, 256, 0, stream>>>(aout, uout, gout, u, v, out);
  (void)in_sizes; (void)n_in; (void)out_size; (void)ws_size;
}
